// I64Dynamics_25323127177574
// MI455X (gfx1250) — compile-verified, matrix operand order checked by reading, not run on hardware
//
#include <hip/hip_runtime.h>
#include <math.h>

// Problem sizes (fixed by the reference)
#define HSZ 2048          // hidden
#define CSZ 256           // controller
#define BSZ 8192          // batch tokens
#define KIN 4096          // 2*H
#define DT_C 0.1f

typedef __attribute__((ext_vector_type(16))) __bf16 v16bf;
typedef __attribute__((ext_vector_type(8)))  float  v8f;

union FragU { v16bf v; uint4 u[2]; };
union Pack4 { __bf16 b[4]; uint2 u; };

__device__ __forceinline__ v8f wmma_bf16(v16bf a, v16bf b, v8f c) {
  return __builtin_amdgcn_wmma_f32_16x16x32_bf16(false, a, false, b,
                                                 (short)0, c, false, false);
}

// fast sigmoid: v_exp_f32 + v_rcp_f32 (no IEEE div sequence)
__device__ __forceinline__ float fast_sigmoid(float x) {
  return __builtin_amdgcn_rcpf(1.0f + __expf(-x));
}

// Lane-local fragment load, CDNA5 ISA 7.12.2 wave32 16x32 bf16 layout.
// p must already point at (row/col + lane&15)*ld + (lane>>4)*8; OFF is a
// compile-time element offset so it folds into the load's IOFFSET.
#define LOAD_FRAG(dstfrag, p, OFF)                                  \
  { (dstfrag).u[0] = *(const uint4*)((p) + (OFF));                  \
    (dstfrag).u[1] = *(const uint4*)((p) + (OFF) + 16); }

// ---------------------------------------------------------------- converts --
// 4 floats -> 4 bf16 per thread (b128 in, b64 out)
__global__ void __launch_bounds__(256)
cvt_f32_bf16_x4(const float4* __restrict__ src, uint2* __restrict__ dst, int n4) {
  int i = blockIdx.x * 256 + threadIdx.x;
  if (i < n4) {
    float4 s = src[i];
    Pack4 p;
    p.b[0] = (__bf16)s.x; p.b[1] = (__bf16)s.y;
    p.b[2] = (__bf16)s.z; p.b[3] = (__bf16)s.w;
    dst[i] = p.u;
  }
}

// -------------------------------------------------- stage 1: ctrl = silu(..)
// grid: (BSZ/16, 1), block 256: 8 waves x (16x32) tiles cover all C=256 cols.
__global__ void __launch_bounds__(256)
ctrl_gemm_silu(const __bf16* __restrict__ hbf,   // [B, H] bf16
               const __bf16* __restrict__ vbf,   // [B, H] bf16
               const __bf16* __restrict__ Win,   // [C, 2H] bf16
               const float*  __restrict__ b_in,  // [C]
               __bf16* __restrict__ ctrl) {      // [B, C] bf16 out
  const int lane = threadIdx.x & 31;
  const int wave = threadIdx.x >> 5;
  const int m0 = blockIdx.x << 4;
  const int n0 = wave << 5;                       // 0..224
  const int kh = (lane >> 4) << 3;
  const int ln = lane & 15;

  const __bf16* pa = hbf + (size_t)(m0 + ln) * HSZ + kh;
  const __bf16* pb = Win + (size_t)(n0 + ln) * KIN + kh;   // advances over all K=4096

  v8f acc0 = {}, acc1 = {};
#pragma unroll 2
  for (int k = 0; k < HSZ; k += 32) {             // hv first half: A from h
    FragU a, b0, b1;
    LOAD_FRAG(a, pa, 0);
    LOAD_FRAG(b0, pb, 0);
    LOAD_FRAG(b1, pb, 16 * KIN);                  // col subtile +16 (imm offset)
    acc0 = wmma_bf16(a.v, b0.v, acc0);
    acc1 = wmma_bf16(a.v, b1.v, acc1);
    pa += 32; pb += 32;
  }
  pa = vbf + (size_t)(m0 + ln) * HSZ + kh;        // hv second half: A from v
#pragma unroll 2
  for (int k = 0; k < HSZ; k += 32) {
    FragU a, b0, b1;
    LOAD_FRAG(a, pa, 0);
    LOAD_FRAG(b0, pb, 0);
    LOAD_FRAG(b1, pb, 16 * KIN);
    acc0 = wmma_bf16(a.v, b0.v, acc0);
    acc1 = wmma_bf16(a.v, b1.v, acc1);
    pa += 32; pb += 32;
  }

  const int mo = (lane >> 4) << 3;
#pragma unroll
  for (int s = 0; s < 2; ++s) {
    v8f acc = s ? acc1 : acc0;
    const int n = n0 + (s << 4) + ln;
    const float bias = b_in[n];
#pragma unroll
    for (int r = 0; r < 8; ++r) {
      float x = acc[r] + bias;
      float sv = x * fast_sigmoid(x);             // SiLU
      ctrl[(size_t)(m0 + mo + r) * CSZ + n] = (__bf16)sv;
    }
  }
}

// ------------------------ per-subtile gate GEMM (K=256) + dynamics epilogue
__device__ __forceinline__ void gates_and_update(
    const __bf16* __restrict__ pca,     // ctrl A base (lane-adjusted)
    const __bf16* __restrict__ Wout,    // [3H, C] bf16
    int n_sub, int kh, int lane, int m0, v8f accmu,
    const float* __restrict__ hf, const float* __restrict__ vf,
    const float* __restrict__ b_out, const float* __restrict__ mu,
    float* __restrict__ out) {
  const int ln = lane & 15;
  const __bf16* pwb = Wout + (size_t)(n_sub + ln) * CSZ + kh;
  v8f aa = {}, ab = {}, ag = {};
#pragma unroll
  for (int k0 = 0; k0 < CSZ; k0 += 32) {
    FragU a, ba, bb, bg;
    LOAD_FRAG(a,  pca, k0);
    LOAD_FRAG(ba, pwb, k0);
    LOAD_FRAG(bb, pwb, k0 + HSZ * CSZ);           // beta rows (+H), imm offset
    LOAD_FRAG(bg, pwb, k0 + 2 * HSZ * CSZ);       // gate rows (+2H), imm offset
    aa = wmma_bf16(a.v, ba.v, aa);
    ab = wmma_bf16(a.v, bb.v, ab);
    ag = wmma_bf16(a.v, bg.v, ag);
  }

  const int n  = n_sub + ln;
  const int mo = (lane >> 4) << 3;
  const float boa = b_out[n];
  const float bob = b_out[HSZ + n];
  const float bog = b_out[2 * HSZ + n];
  const float mun = mu[n];
  float* out_h = out;
  float* out_v = out + (size_t)BSZ * HSZ;
  float* out_m = out + 2 * (size_t)BSZ * HSZ;

#pragma unroll
  for (int r = 0; r < 8; ++r) {
    const size_t idx = (size_t)(m0 + mo + r) * HSZ + n;
    float araw = aa[r] + boa;
    float braw = ab[r] + bob;
    float graw = ag[r] + bog;
    float alpha = fast_sigmoid(araw);
    // stable softplus via v_exp/v_log, capped at 2
    float sp   = fmaxf(braw, 0.0f) + __logf(1.0f + __expf(-fabsf(braw)));
    float beta = fminf(sp, 2.0f);
    float gate = fast_sigmoid(graw);
    float muc  = mun + accmu[r];
    float hval = hf[idx];
    float vval = vf[idx];
    float err  = hval - muc;
    float vn   = fminf(fmaxf(alpha * vval - beta * err, -10.0f), 10.0f);
    float hn   = fmaf(DT_C * gate, vn, hval);
    out_h[idx] = hn;
    out_v[idx] = vn;
    out_m[idx] = muc;
  }
}

// --------------------------------- stage 2+3+4 fused: mu GEMM + gates + update
// grid: (BSZ/16, HSZ/512), block 256: 8 waves x (16x64) strips.
__global__ void __launch_bounds__(256)
fused_dynamics(const __bf16* __restrict__ hbf,    // [B, H] bf16
               const __bf16* __restrict__ ctrl,   // [B, C] bf16
               const __bf16* __restrict__ Wout,   // [3H, C] bf16
               const __bf16* __restrict__ Wmu,    // [H, H] bf16
               const float*  __restrict__ hf,     // [B, H] f32 original
               const float*  __restrict__ vf,     // [B, H] f32 original
               const float*  __restrict__ b_out,  // [3H]
               const float*  __restrict__ mu,     // [H]
               float* __restrict__ out) {         // h_next | v_next | mu_ctx
  const int lane = threadIdx.x & 31;
  const int wave = threadIdx.x >> 5;
  const int m0 = blockIdx.x << 4;
  const int n0 = (blockIdx.y << 9) + (wave << 6); // 64-wide strip per wave
  const int kh = (lane >> 4) << 3;
  const int ln = lane & 15;

  // --- mu_ctx strip: h @ W_mu^T, K = 2048, A frag shared by 4 subtiles ---
  const __bf16* pa = hbf + (size_t)(m0 + ln) * HSZ + kh;
  const __bf16* pb = Wmu + (size_t)(n0 + ln) * HSZ + kh;
  v8f amu0 = {}, amu1 = {}, amu2 = {}, amu3 = {};
#pragma unroll 2
  for (int k0 = 0; k0 < HSZ; k0 += 32) {
    FragU a, b0, b1, b2, b3;
    LOAD_FRAG(a,  pa, 0);
    LOAD_FRAG(b0, pb, 0);
    LOAD_FRAG(b1, pb, 16 * HSZ);                  // subtile col offsets fold
    LOAD_FRAG(b2, pb, 32 * HSZ);                  // into the load IOFFSET
    LOAD_FRAG(b3, pb, 48 * HSZ);
    amu0 = wmma_bf16(a.v, b0.v, amu0);
    amu1 = wmma_bf16(a.v, b1.v, amu1);
    amu2 = wmma_bf16(a.v, b2.v, amu2);
    amu3 = wmma_bf16(a.v, b3.v, amu3);
    __builtin_prefetch(pa + 512, 0, 1);           // global_prefetch_b8
    __builtin_prefetch(pb + 512, 0, 1);
    pa += 32; pb += 32;
  }

  // --- gate GEMMs (K=256) + epilogue, one 16x16 subtile at a time ---
  const __bf16* pca = ctrl + (size_t)(m0 + ln) * CSZ + kh;
  gates_and_update(pca, Wout, n0 +  0, kh, lane, m0, amu0, hf, vf, b_out, mu, out);
  gates_and_update(pca, Wout, n0 + 16, kh, lane, m0, amu1, hf, vf, b_out, mu, out);
  gates_and_update(pca, Wout, n0 + 32, kh, lane, m0, amu2, hf, vf, b_out, mu, out);
  gates_and_update(pca, Wout, n0 + 48, kh, lane, m0, amu3, hf, vf, b_out, mu, out);
}

// ---------------------------------------------------------------- launcher --
extern "C" void kernel_launch(void* const* d_in, const int* in_sizes, int n_in,
                              void* d_out, int out_size, void* d_ws, size_t ws_size,
                              hipStream_t stream) {
  const float* h     = (const float*)d_in[0];
  const float* v     = (const float*)d_in[1];
  const float* W_in  = (const float*)d_in[2];
  const float* b_in  = (const float*)d_in[3];
  const float* W_out = (const float*)d_in[4];
  const float* b_out = (const float*)d_in[5];
  const float* W_mu  = (const float*)d_in[6];
  const float* mu    = (const float*)d_in[7];
  float* out = (float*)d_out;

  // workspace layout (bf16 staging, ~84 MB total)
  char* ws = (char*)d_ws;
  const size_t nBH   = (size_t)BSZ * HSZ;        // 16.7M
  const size_t nWin  = (size_t)CSZ * KIN;        // 1.05M
  const size_t nWout = (size_t)(3 * HSZ) * CSZ;  // 1.57M
  const size_t nWmu  = (size_t)HSZ * HSZ;        // 4.19M
  const size_t nCtrl = (size_t)BSZ * CSZ;        // 2.10M
  __bf16* hbf    = (__bf16*)ws;                     ws += nBH   * 2;
  __bf16* vbf    = (__bf16*)ws;                     ws += nBH   * 2;
  __bf16* Winbf  = (__bf16*)ws;                     ws += nWin  * 2;
  __bf16* Woutbf = (__bf16*)ws;                     ws += nWout * 2;
  __bf16* Wmubf  = (__bf16*)ws;                     ws += nWmu  * 2;
  __bf16* ctrlbf = (__bf16*)ws;                     ws += nCtrl * 2;

  // 1) converts to bf16 (4 elems/thread)
  cvt_f32_bf16_x4<<<(int)((nBH/4   + 255) / 256), 256, 0, stream>>>((const float4*)h,     (uint2*)hbf,    (int)(nBH/4));
  cvt_f32_bf16_x4<<<(int)((nBH/4   + 255) / 256), 256, 0, stream>>>((const float4*)v,     (uint2*)vbf,    (int)(nBH/4));
  cvt_f32_bf16_x4<<<(int)((nWin/4  + 255) / 256), 256, 0, stream>>>((const float4*)W_in,  (uint2*)Winbf,  (int)(nWin/4));
  cvt_f32_bf16_x4<<<(int)((nWout/4 + 255) / 256), 256, 0, stream>>>((const float4*)W_out, (uint2*)Woutbf, (int)(nWout/4));
  cvt_f32_bf16_x4<<<(int)((nWmu/4  + 255) / 256), 256, 0, stream>>>((const float4*)W_mu,  (uint2*)Wmubf,  (int)(nWmu/4));

  // 2) ctrl = silu([h,v] @ W_in^T + b_in)   -> bf16 [B, C]
  ctrl_gemm_silu<<<dim3(BSZ / 16, 1), 256, 0, stream>>>(
      hbf, vbf, Winbf, b_in, ctrlbf);

  // 3) fused mu GEMM + gate GEMMs + dynamics update -> 3 f32 outputs
  fused_dynamics<<<dim3(BSZ / 16, HSZ / 512), 256, 0, stream>>>(
      hbf, ctrlbf, Woutbf, Wmubf, h, v, b_out, mu, out);
}
